// PCT_Tokenizer_59356448031601
// MI455X (gfx1250) — compile-verified, matrix operand order checked by reading, not run on hardware
//
#include <hip/hip_runtime.h>
#include <hip/hip_bf16.h>
#include <stdint.h>

// ============================================================================
// PCT tokenizer for MI455X (gfx1250, wave32).
//  - All K=512 GEMMs (channel MLPs, feat_embed, VQ) -> v_wmma_f32_16x16x32_bf16
//  - GEMM K-loop: double-buffered LDS tiles fed by GLOBAL_LOAD_ASYNC_TO_LDS_B128
//    (ASYNCcnt) when the toolchain exposes the builtin; VGPR-staged fallback
//    otherwise. Async path overlaps tile k+1 fetch with tile k WMMAs and skips
//    the VGPR round-trip.
//  - VQ argmin fused into GEMM epilogue via packed u64 atomicMin (avoids
//    materializing the 570MB distance matrix -> saves ~50us of HBM traffic)
//  - K=17 token mixers / decoder (H=32) -> fused VALU kernels (too skinny for
//    16x16 tiles)
// ============================================================================

typedef __bf16 bf16_t;
typedef __attribute__((ext_vector_type(16))) __bf16 v16bf;
typedef __attribute__((ext_vector_type(8)))  float  v8f;

#define DEV __device__ __forceinline__

#if __has_builtin(__builtin_amdgcn_global_load_async_to_lds_b128) && \
    __has_builtin(__builtin_amdgcn_s_wait_asynccnt)
#define ASYNC_LDS 1
#else
#define ASYNC_LDS 0
#endif

#if ASYNC_LDS
// builtin expects pointers to 16B int vectors (per clang diagnostic);
// global src in addrspace(1), LDS dst in addrspace(3)
typedef int v4i_t __attribute__((vector_size(16)));
typedef __attribute__((address_space(1))) v4i_t gv4i_t;
typedef __attribute__((address_space(3))) v4i_t lv4i_t;
DEV void async_ld16(const void* g, void* l) {
    // 16B global -> LDS, no VGPR staging; tracked by ASYNCcnt
    __builtin_amdgcn_global_load_async_to_lds_b128((gv4i_t*)g, (lv4i_t*)l, 0, 0);
}
#endif

DEV float geluf(float x) { return 0.5f * x * (1.0f + erff(x * 0.7071067811865476f)); }

// order-preserving f32 -> u32 (handles negatives)
DEV unsigned fkey(float f) {
    unsigned u = __float_as_uint(f);
    return (u & 0x80000000u) ? ~u : (u | 0x80000000u);
}

DEV unsigned long long shflxor64(unsigned long long v, int m) {
    unsigned lo = (unsigned)v, hi = (unsigned)(v >> 32);
    lo = __shfl_xor(lo, m, 32);
    hi = __shfl_xor(hi, m, 32);
    return ((unsigned long long)hi << 32) | lo;
}

// ---------------------------------------------------------------------------
// WMMA GEMM:  C[M,N] = A[M,K](bf16) * BT[N,K](bf16)^T  (+ epilogue)
// 256 threads = 8 waves; block tile 128x128; wave tile 32x64 (2x4 WMMA accs);
// BK=32 matches v_wmma_f32_16x16x32_bf16. LDS double-buffered (40KB).
// ---------------------------------------------------------------------------
enum { EPI_BIAS_F32 = 0, EPI_GELU_BF16 = 1, EPI_BIAS_BF16 = 2, EPI_RES_F32 = 3, EPI_VQ = 4 };

template <int EPI>
__global__ __launch_bounds__(256) void gemm_bf16(
    const bf16_t* __restrict__ A, const bf16_t* __restrict__ BT,
    const float* __restrict__ bias, const float* __restrict__ res,
    float* __restrict__ outF, bf16_t* __restrict__ outB,
    const float* __restrict__ cbnorm, unsigned long long* __restrict__ argws,
    int M, int N, int Kd)
{
    __shared__ bf16_t As[2][128][40];   // BK=32 + 8 pad (80B stride, 16B aligned)
    __shared__ bf16_t Bs[2][128][40];   // n-major (BT layout): row = n, col = k

    const int tid = threadIdx.x, lane = tid & 31, wid = tid >> 5;
    const int wm = wid & 3, wn = wid >> 2;           // wave grid 4(M) x 2(N)
    const int m0 = blockIdx.y * 128, n0 = blockIdx.x * 128;
    const int lrow = tid >> 2, lcol = (tid & 3) * 8; // 4 threads x 8 bf16 per row

    v8f acc[2][4];
#pragma unroll
    for (int i = 0; i < 2; i++)
#pragma unroll
        for (int j = 0; j < 4; j++)
#pragma unroll
            for (int r = 0; r < 8; r++) acc[i][j][r] = 0.f;

    const int half = (lane >> 4) * 8, l16 = lane & 15;

#if ASYNC_LDS
    // ---- async double-buffered pipeline ----
    const int amA = m0 + lrow, amB = m0 + 64 + lrow;
    const int bnA = n0 + lrow, bnB = n0 + 64 + lrow;

    // OOB rows are K-invariant: zero them in LDS once; async loads skip them.
    {
        const uint4 z4 = {0, 0, 0, 0};
#pragma unroll
        for (int p = 0; p < 2; p++) {
            int r = p * 64 + lrow;
            if (m0 + r >= M) { *(uint4*)&As[0][r][lcol] = z4; *(uint4*)&As[1][r][lcol] = z4; }
            if (n0 + r >= N) { *(uint4*)&Bs[0][r][lcol] = z4; *(uint4*)&Bs[1][r][lcol] = z4; }
        }
    }
    // prologue: kick off tile 0
    {
        if (amA < M) async_ld16(A + (size_t)amA * Kd + lcol, &As[0][lrow][lcol]);
        if (amB < M) async_ld16(A + (size_t)amB * Kd + lcol, &As[0][64 + lrow][lcol]);
        if (bnA < N) async_ld16(BT + (size_t)bnA * Kd + lcol, &Bs[0][lrow][lcol]);
        if (bnB < N) async_ld16(BT + (size_t)bnB * Kd + lcol, &Bs[0][64 + lrow][lcol]);
    }
    const int nk = Kd >> 5;
    for (int kt = 0; kt < nk; kt++) {
        const int cur = kt & 1;
        __builtin_amdgcn_s_wait_asynccnt(0);  // tile kt resident in LDS
        __syncthreads();
        if (kt + 1 < nk) {                    // overlap: fetch tile kt+1
            const int kn = (kt + 1) * 32 + lcol;
            if (amA < M) async_ld16(A + (size_t)amA * Kd + kn, &As[cur ^ 1][lrow][lcol]);
            if (amB < M) async_ld16(A + (size_t)amB * Kd + kn, &As[cur ^ 1][64 + lrow][lcol]);
            if (bnA < N) async_ld16(BT + (size_t)bnA * Kd + kn, &Bs[cur ^ 1][lrow][lcol]);
            if (bnB < N) async_ld16(BT + (size_t)bnB * Kd + kn, &Bs[cur ^ 1][64 + lrow][lcol]);
        }
        // fragments per ISA 16-bit A layout:
        // lanes 0-15: rows, K 0..7 & 16..23 ; lanes 16-31: K 8..15 & 24..31
        union Frag { v16bf v; uint4 q[2]; } fa[2], fb[4];
#pragma unroll
        for (int i = 0; i < 2; i++) {
            int rr = wm * 32 + i * 16 + l16;
            fa[i].q[0] = *(const uint4*)&As[cur][rr][half];
            fa[i].q[1] = *(const uint4*)&As[cur][rr][16 + half];
        }
#pragma unroll
        for (int j = 0; j < 4; j++) {
            int rr = wn * 64 + j * 16 + l16;
            fb[j].q[0] = *(const uint4*)&Bs[cur][rr][half];
            fb[j].q[1] = *(const uint4*)&Bs[cur][rr][16 + half];
        }
#pragma unroll
        for (int i = 0; i < 2; i++)
#pragma unroll
            for (int j = 0; j < 4; j++)
                acc[i][j] = __builtin_amdgcn_wmma_f32_16x16x32_bf16(
                    false, fa[i].v, false, fb[j].v, (short)0, acc[i][j], false, false);
        // no trailing barrier: next iteration's wait+barrier fences buffer reuse
    }
#else
    // ---- fallback: VGPR-staged single-buffer pipeline ----
    for (int k0 = 0; k0 < Kd; k0 += 32) {
#pragma unroll
        for (int p = 0; p < 2; p++) {
            int r = p * 64 + lrow;
            uint4 va = {0, 0, 0, 0}, vb = {0, 0, 0, 0};
            int am = m0 + r;
            if (am < M) va = *(const uint4*)(A + (size_t)am * Kd + k0 + lcol);
            int bn = n0 + r;
            if (bn < N) vb = *(const uint4*)(BT + (size_t)bn * Kd + k0 + lcol);
            *(uint4*)&As[0][r][lcol] = va;
            *(uint4*)&Bs[0][r][lcol] = vb;
        }
        if (k0 + 32 < Kd) {  // emits global_prefetch_b8
            int am = m0 + lrow;
            if (am < M) __builtin_prefetch(A + (size_t)am * Kd + k0 + 32, 0, 1);
            int bn = n0 + lrow;
            if (bn < N) __builtin_prefetch(BT + (size_t)bn * Kd + k0 + 32, 0, 1);
        }
        __syncthreads();
        union Frag { v16bf v; uint4 q[2]; } fa[2], fb[4];
#pragma unroll
        for (int i = 0; i < 2; i++) {
            int rr = wm * 32 + i * 16 + l16;
            fa[i].q[0] = *(const uint4*)&As[0][rr][half];
            fa[i].q[1] = *(const uint4*)&As[0][rr][16 + half];
        }
#pragma unroll
        for (int j = 0; j < 4; j++) {
            int rr = wn * 64 + j * 16 + l16;
            fb[j].q[0] = *(const uint4*)&Bs[0][rr][half];
            fb[j].q[1] = *(const uint4*)&Bs[0][rr][16 + half];
        }
#pragma unroll
        for (int i = 0; i < 2; i++)
#pragma unroll
            for (int j = 0; j < 4; j++)
                acc[i][j] = __builtin_amdgcn_wmma_f32_16x16x32_bf16(
                    false, fa[i].v, false, fb[j].v, (short)0, acc[i][j], false, false);
        __syncthreads();
    }
#endif

    // ---- epilogue ----
    const int rb8 = (lane >> 4) * 8;
    if constexpr (EPI == EPI_VQ) {
        // score = |cb_n|^2 - 2*dot  (row-constant |flat|^2 dropped: argmin-invariant)
#pragma unroll
        for (int i = 0; i < 2; i++) {
#pragma unroll
            for (int r = 0; r < 8; r++) {
                int m = m0 + wm * 32 + i * 16 + rb8 + r;
                unsigned long long best = ~0ull;
                if (m < M) {
#pragma unroll
                    for (int j = 0; j < 4; j++) {
                        int n = n0 + wn * 64 + j * 16 + l16;
                        if (n < N) {
                            float sc = cbnorm[n] - 2.f * acc[i][j][r];
                            unsigned long long key =
                                ((unsigned long long)fkey(sc) << 32) | (unsigned)n;
                            if (key < best) best = key;
                        }
                    }
                }
                for (int mk = 8; mk; mk >>= 1) {
                    unsigned long long o = shflxor64(best, mk);
                    if (o < best) best = o;
                }
                if (l16 == 0 && m < M) atomicMin(&argws[m], best);
            }
        }
    } else {
#pragma unroll
        for (int i = 0; i < 2; i++)
#pragma unroll
            for (int j = 0; j < 4; j++) {
                int n = n0 + wn * 64 + j * 16 + l16;
#pragma unroll
                for (int r = 0; r < 8; r++) {
                    int m = m0 + wm * 32 + i * 16 + rb8 + r;
                    if (m < M && n < N) {
                        float v = acc[i][j][r] + bias[n];
                        size_t o = (size_t)m * N + n;
                        if constexpr (EPI == EPI_BIAS_F32)  outF[o] = v;
                        if constexpr (EPI == EPI_GELU_BF16) outB[o] = (bf16_t)geluf(v);
                        if constexpr (EPI == EPI_BIAS_BF16) outB[o] = (bf16_t)v;
                        if constexpr (EPI == EPI_RES_F32)   outF[o] = res[o] + v;
                    }
                }
            }
    }
}

// ---------------------------------------------------------------------------
// LayerNorm over last dim (H), one block per row; f32 and/or bf16 outputs.
// ---------------------------------------------------------------------------
__global__ __launch_bounds__(256) void k_layernorm(
    const float* __restrict__ x, const float* __restrict__ g, const float* __restrict__ b,
    float* __restrict__ outF, bf16_t* __restrict__ outB, int H)
{
    size_t row = blockIdx.x;
    const float* xr = x + row * H;
    __shared__ float red[8];
    int tid = threadIdx.x, lane = tid & 31, wid = tid >> 5;

    float s = 0.f;
    for (int i = tid; i < H; i += 256) s += xr[i];
    for (int m = 16; m; m >>= 1) s += __shfl_xor(s, m, 32);
    if (lane == 0) red[wid] = s;
    __syncthreads();
    float mu = 0.f;
    for (int w = 0; w < 8; w++) mu += red[w];
    mu /= (float)H;
    __syncthreads();

    float v = 0.f;
    for (int i = tid; i < H; i += 256) { float d = xr[i] - mu; v += d * d; }
    for (int m = 16; m; m >>= 1) v += __shfl_xor(v, m, 32);
    if (lane == 0) red[wid] = v;
    __syncthreads();
    float var = 0.f;
    for (int w = 0; w < 8; w++) var += red[w];
    var /= (float)H;
    float rs = rsqrtf(var + 1e-5f);

    for (int i = tid; i < H; i += 256) {
        float y = (xr[i] - mu) * rs * g[i] + b[i];
        if (outF) outF[row * H + i] = y;
        if (outB) outB[row * H + i] = (bf16_t)y;
    }
}

// ---------------------------------------------------------------------------
// Start embed (2->512) + visibility masking with invisible token.
// ---------------------------------------------------------------------------
__global__ __launch_bounds__(256) void k_embed_mask(
    const float* __restrict__ fm, const float* __restrict__ joints,
    const float* __restrict__ rnd, const float* __restrict__ wse,
    const float* __restrict__ bse, const float* __restrict__ inv,
    float* __restrict__ X)
{
    int t = blockIdx.x * 256 + threadIdx.x;
    if (t >= 34816 * 512) return;
    int bk = t >> 9, h = t & 511;
    int b_ = bk / 17, k_ = bk % 17;
    const float* f0 = fm + (((size_t)b_ * 2 + 0) * 17 + k_) * 2;
    const float* f1 = fm + (((size_t)b_ * 2 + 1) * 17 + k_) * 2;
    float m0 = 0.5f * (f0[0] + f1[0]), m1 = 0.5f * (f0[1] + f1[1]);
    float x = m0 * wse[h] + m1 * wse[512 + h] + bse[h];
    bool vis = (rnd[bk] > 0.2f) && (joints[(size_t)bk * 3 + 2] != 0.f);
    X[t] = vis ? x : inv[h];
}

// ---------------------------------------------------------------------------
// Token-mixing MLP across joints (K->TI->K), weights in LDS, one thread per
// (batch, channel).  Writes xy = x + tokmix(ln).
// ---------------------------------------------------------------------------
template <int KJ, int TI>
__global__ __launch_bounds__(256) void k_token_mix(
    const float* __restrict__ ln, const float* __restrict__ xres,
    const float* __restrict__ w1, const float* __restrict__ b1,
    const float* __restrict__ w2, const float* __restrict__ b2,
    float* __restrict__ xy, int B, int Hdim)
{
    __shared__ float sw1[KJ * TI], sw2[TI * KJ], sb1[TI], sb2[KJ];
    for (int i = threadIdx.x; i < KJ * TI; i += 256) sw1[i] = w1[i];
    for (int i = threadIdx.x; i < TI * KJ; i += 256) sw2[i] = w2[i];
    if (threadIdx.x < TI) sb1[threadIdx.x] = b1[threadIdx.x];
    if (threadIdx.x < KJ) sb2[threadIdx.x] = b2[threadIdx.x];
    __syncthreads();
    int t = blockIdx.x * 256 + threadIdx.x;
    if (t >= B * Hdim) return;
    int b_ = t / Hdim, h = t - b_ * Hdim;
    float xv[KJ], acc[KJ];
#pragma unroll
    for (int k = 0; k < KJ; k++) { xv[k] = ln[((size_t)b_ * KJ + k) * Hdim + h]; acc[k] = 0.f; }
    for (int i = 0; i < TI; i++) {
        float s = sb1[i];
#pragma unroll
        for (int k = 0; k < KJ; k++) s += xv[k] * sw1[k * TI + i];
        float g = geluf(s);
#pragma unroll
        for (int k = 0; k < KJ; k++) acc[k] += g * sw2[i * KJ + k];
    }
#pragma unroll
    for (int k = 0; k < KJ; k++) {
        size_t o = ((size_t)b_ * KJ + k) * Hdim + h;
        xy[o] = xres[o] + acc[k] + sb2[k];
    }
}

// ---------------------------------------------------------------------------
// Joints->tokens projection (K -> MO, no gelu), bf16 output rows [B*MO, H].
// ---------------------------------------------------------------------------
template <int KJ, int MO>
__global__ __launch_bounds__(256) void k_token_mix_km(
    const float* __restrict__ ln, const float* __restrict__ w,
    const float* __restrict__ bias, bf16_t* __restrict__ outB, int B, int Hdim)
{
    __shared__ float sw[KJ * MO], sb[MO];
    for (int i = threadIdx.x; i < KJ * MO; i += 256) sw[i] = w[i];
    if (threadIdx.x < MO) sb[threadIdx.x] = bias[threadIdx.x];
    __syncthreads();
    int t = blockIdx.x * 256 + threadIdx.x;
    if (t >= B * Hdim) return;
    int b_ = t / Hdim, h = t - b_ * Hdim;
    float xv[KJ];
#pragma unroll
    for (int k = 0; k < KJ; k++) xv[k] = ln[((size_t)b_ * KJ + k) * Hdim + h];
    for (int m = 0; m < MO; m++) {
        float s = sb[m];
#pragma unroll
        for (int k = 0; k < KJ; k++) s += xv[k] * sw[k * MO + m];
        outB[((size_t)b_ * MO + m) * Hdim + h] = (bf16_t)s;
    }
}

// ---------------------------------------------------------------------------
// Weight convert+transpose: W f32 [Din,Dout] -> WT bf16 [Dout,Din].
// ---------------------------------------------------------------------------
__global__ __launch_bounds__(256) void k_wt(
    const float* __restrict__ W, bf16_t* __restrict__ WT, int Din, int Dout)
{
    int t = blockIdx.x * 256 + threadIdx.x;
    if (t >= Din * Dout) return;
    int n = t / Din, k = t - n * Din;
    WT[t] = (bf16_t)W[(size_t)k * Dout + n];
}

// codebook: bf16 copy (already [N][K] layout) + per-row squared norms
__global__ __launch_bounds__(256) void k_cb_prep(
    const float* __restrict__ cb, bf16_t* __restrict__ cbbf, float* __restrict__ cbn)
{
    int c = blockIdx.x, tid = threadIdx.x, lane = tid & 31, wid = tid >> 5;
    __shared__ float red[8];
    float s = 0.f;
    for (int i = tid; i < 512; i += 256) {
        float v = cb[(size_t)c * 512 + i];
        cbbf[(size_t)c * 512 + i] = (bf16_t)v;
        s += v * v;
    }
    for (int m = 16; m; m >>= 1) s += __shfl_xor(s, m, 32);
    if (lane == 0) red[wid] = s;
    __syncthreads();
    if (tid == 0) {
        float tot = 0.f;
        for (int w = 0; w < 8; w++) tot += red[w];
        cbn[c] = tot;
    }
}

__global__ __launch_bounds__(256) void k_init64(unsigned long long* p, int n)
{
    int t = blockIdx.x * 256 + threadIdx.x;
    if (t < n) p[t] = ~0ull;
}

__global__ __launch_bounds__(256) void k_idx_out(
    const unsigned long long* __restrict__ argws, int* __restrict__ out, int n)
{
    int t = blockIdx.x * 256 + threadIdx.x;
    if (t < n) out[t] = (int)(argws[t] & 0xffffffffull);
}

// ---------------------------------------------------------------------------
// Decoder: gather codebook rows via argmin indices, mix tokens->joints (34->17),
// bf16 rows [B*17, 512] for the dec_start GEMM.
// ---------------------------------------------------------------------------
__global__ __launch_bounds__(256) void k_dec_token_mix(
    const float* __restrict__ cb, const unsigned long long* __restrict__ argws,
    const float* __restrict__ w, const float* __restrict__ bias,
    bf16_t* __restrict__ pjbf)
{
    __shared__ float sw[34 * 17], sb[17];
    for (int i = threadIdx.x; i < 34 * 17; i += 256) sw[i] = w[i];
    if (threadIdx.x < 17) sb[threadIdx.x] = bias[threadIdx.x];
    __syncthreads();
    int t = blockIdx.x * 256 + threadIdx.x;
    if (t >= 2048 * 512) return;
    int b_ = t >> 9, d = t & 511;
    float pv[34];
#pragma unroll
    for (int m = 0; m < 34; m++) {
        int idx = (int)(argws[(size_t)b_ * 34 + m] & 0xffffffffull);
        pv[m] = cb[(size_t)idx * 512 + d];
    }
#pragma unroll
    for (int k = 0; k < 17; k++) {
        float s = sb[k];
#pragma unroll
        for (int m = 0; m < 34; m++) s += pv[m] * sw[m * 17 + k];
        pjbf[((size_t)b_ * 17 + k) * 512 + d] = (bf16_t)s;
    }
}

// Decoder channel MLP (32->64->32) with residual, one thread per row.
__global__ __launch_bounds__(256) void k_dec_channel(
    const float* __restrict__ z, const float* __restrict__ xy,
    const float* __restrict__ w1, const float* __restrict__ b1,
    const float* __restrict__ w2, const float* __restrict__ b2,
    float* __restrict__ out, int rows)
{
    __shared__ float sw1[32 * 64], sw2[64 * 32], sb1[64], sb2[32];
    for (int i = threadIdx.x; i < 2048; i += 256) { sw1[i] = w1[i]; sw2[i] = w2[i]; }
    if (threadIdx.x < 64) sb1[threadIdx.x] = b1[threadIdx.x];
    if (threadIdx.x < 32) sb2[threadIdx.x] = b2[threadIdx.x];
    __syncthreads();
    int t = blockIdx.x * 256 + threadIdx.x;
    if (t >= rows) return;
    const float* zr = z + (size_t)t * 32;
    float zi[32], acc[32];
#pragma unroll
    for (int j = 0; j < 32; j++) { zi[j] = zr[j]; acc[j] = 0.f; }
    for (int i = 0; i < 64; i++) {
        float s = sb1[i];
#pragma unroll
        for (int j = 0; j < 32; j++) s += zi[j] * sw1[j * 64 + i];
        float g = geluf(s);
#pragma unroll
        for (int j = 0; j < 32; j++) acc[j] += g * sw2[i * 32 + j];
    }
#pragma unroll
    for (int j = 0; j < 32; j++) {
        size_t o = (size_t)t * 32 + j;
        out[o] = xy[o] + acc[j] + sb2[j];
    }
}

// Final LN(32) + recover (32->2), one thread per (b,k).
__global__ __launch_bounds__(256) void k_recover(
    const float* __restrict__ dec, const float* __restrict__ g, const float* __restrict__ b,
    const float* __restrict__ rw, const float* __restrict__ rb,
    float* __restrict__ out, int rows)
{
    int t = blockIdx.x * 256 + threadIdx.x;
    if (t >= rows) return;
    const float* xr = dec + (size_t)t * 32;
    float mu = 0.f;
#pragma unroll
    for (int i = 0; i < 32; i++) mu += xr[i];
    mu *= (1.f / 32.f);
    float var = 0.f;
#pragma unroll
    for (int i = 0; i < 32; i++) { float d = xr[i] - mu; var += d * d; }
    var *= (1.f / 32.f);
    float rs = rsqrtf(var + 1e-5f);
    float o0 = rb[0], o1 = rb[1];
#pragma unroll
    for (int i = 0; i < 32; i++) {
        float y = (xr[i] - mu) * rs * g[i] + b[i];
        o0 += y * rw[i * 2];
        o1 += y * rw[i * 2 + 1];
    }
    out[(size_t)t * 2]     = o0;
    out[(size_t)t * 2 + 1] = o1;
}

// ============================================================================
// Host-side orchestration
// ============================================================================
static inline int cdiv(int a, int b) { return (a + b - 1) / b; }

struct MixerP {
    const float *ch_b1, *ch_b2, *ch_w1, *ch_w2;
    const float *ln1_b, *ln1_g, *ln2_b, *ln2_g;
    const float *tok_b1, *tok_b2, *tok_w1, *tok_w2;
};

extern "C" void kernel_launch(void* const* d_in, const int* in_sizes, int n_in,
                              void* d_out, int out_size, void* d_ws, size_t ws_size,
                              hipStream_t stream)
{
    (void)in_sizes; (void)n_in; (void)out_size; (void)ws_size;

    // --- input mapping: top-level dict insertion order; params pytree-flattened
    //     in sorted-key order (jax convention). ---
    const float* feature_map = (const float*)d_in[0];
    const float* joints      = (const float*)d_in[1];
    const float* rand_vals   = (const float*)d_in[2];
    // d_in[3] cls_logits unused
    auto mixer_at = [&](int base) {
        MixerP m;
        m.ch_b1 = (const float*)d_in[base + 0];  m.ch_b2 = (const float*)d_in[base + 1];
        m.ch_w1 = (const float*)d_in[base + 2];  m.ch_w2 = (const float*)d_in[base + 3];
        m.ln1_b = (const float*)d_in[base + 4];  m.ln1_g = (const float*)d_in[base + 5];
        m.ln2_b = (const float*)d_in[base + 6];  m.ln2_g = (const float*)d_in[base + 7];
        m.tok_b1 = (const float*)d_in[base + 8]; m.tok_b2 = (const float*)d_in[base + 9];
        m.tok_w1 = (const float*)d_in[base + 10]; m.tok_w2 = (const float*)d_in[base + 11];
        return m;
    };
    const float* codebook       = (const float*)d_in[4];
    MixerP dec_mix              = mixer_at(5);
    const float* dec_ln_b       = (const float*)d_in[17];
    const float* dec_ln_g       = (const float*)d_in[18];
    const float* dec_start_b    = (const float*)d_in[19];
    const float* dec_start_w    = (const float*)d_in[20];
    const float* dec_tok_mlp_b  = (const float*)d_in[21];
    const float* dec_tok_mlp_w  = (const float*)d_in[22];
    MixerP enc_mix[4] = { mixer_at(23), mixer_at(35), mixer_at(47), mixer_at(59) };
    const float* enc_ln_b       = (const float*)d_in[71];
    const float* enc_ln_g       = (const float*)d_in[72];
    const float* feat_embed_b   = (const float*)d_in[73];
    const float* feat_embed_w   = (const float*)d_in[74];
    const float* invisible_tok  = (const float*)d_in[75];
    const float* recover_b      = (const float*)d_in[76];
    const float* recover_w      = (const float*)d_in[77];
    const float* start_embed_b  = (const float*)d_in[78];
    const float* start_embed_w  = (const float*)d_in[79];
    const float* token_mlp_b    = (const float*)d_in[80];
    const float* token_mlp_w    = (const float*)d_in[81];

    // --- workspace carve-up (total ~290 MB; lifetime-based aliasing) ---
    const size_t ROWS  = 34816;           // B*K = 2048*17
    const size_t F32_RH = ROWS * 512 * 4; // 71,303,168
    const size_t BF_RH  = ROWS * 512 * 2; // 35,651,584
    char* ws = (char*)d_ws;
    float*  X     = (float*)(ws);                        // activations [ROWS,512]
    float*  XY    = (float*)(ws + F32_RH);               // residual; later TOKBF
    float*  LN1F  = (float*)(ws + 2 * F32_RH);           // ln f32; later FLATBF
    bf16_t* LNBF  = (bf16_t*)(ws + 3 * F32_RH);          // ln2 bf16; later PJBF
    bf16_t* H1BF  = (bf16_t*)(ws + 3 * F32_RH + BF_RH);  // hidden bf16; later DEC bufs
    char*   wseg  = ws + 3 * F32_RH + 2 * BF_RH;
    bf16_t* WT1   = (bf16_t*)(wseg);                     // 512KB
    bf16_t* WT2   = (bf16_t*)(wseg + 524288);            // 512KB
    bf16_t* FEWT  = (bf16_t*)(wseg + 2 * 524288);        // 512KB
    bf16_t* DSWT  = (bf16_t*)(wseg + 3 * 524288);        // 32KB
    bf16_t* CBBF  = (bf16_t*)(wseg + 3 * 524288 + 32768);                 // 2MB
    float*  CBN   = (float*)(wseg + 3 * 524288 + 32768 + 2097152);        // 8KB
    unsigned long long* ARGWS =
        (unsigned long long*)(wseg + 3 * 524288 + 32768 + 2097152 + 8192); // 544KB
    // aliases
    bf16_t* TOKBF  = (bf16_t*)XY;    // [69632,512] bf16 (71MB fits)
    bf16_t* FLATBF = (bf16_t*)LN1F;  // [69632,512] bf16
    bf16_t* PJBF   = LNBF;           // [ROWS,512] bf16
    float*  DEC    = (float*)H1BF;                     // [ROWS,32]
    float*  DECXY  = (float*)((char*)H1BF + 4456448);  // [ROWS,32]
    float*  DECLN  = (float*)((char*)H1BF + 8912896);  // [ROWS,32]

    const dim3 gCH(4, 272);    // N=512,  M=34816
    const dim3 gFE(4, 544);    // N=512,  M=69632
    const dim3 gVQ(16, 544);   // N=2048, M=69632
    const dim3 gDS(1, 272);    // N=32,   M=34816

    // --- 1. start embed + masking ---
    k_embed_mask<<<cdiv(34816 * 512, 256), 256, 0, stream>>>(
        feature_map, joints, rand_vals, start_embed_w, start_embed_b, invisible_tok, X);

    // --- 2. encoder mixer blocks ---
    for (int L = 0; L < 4; L++) {
        const MixerP& m = enc_mix[L];
        k_layernorm<<<(int)ROWS, 256, 0, stream>>>(X, m.ln1_g, m.ln1_b, LN1F, nullptr, 512);
        k_token_mix<17, 64><<<cdiv(2048 * 512, 256), 256, 0, stream>>>(
            LN1F, X, m.tok_w1, m.tok_b1, m.tok_w2, m.tok_b2, XY, 2048, 512);
        k_layernorm<<<(int)ROWS, 256, 0, stream>>>(XY, m.ln2_g, m.ln2_b, nullptr, LNBF, 512);
        k_wt<<<cdiv(512 * 512, 256), 256, 0, stream>>>(m.ch_w1, WT1, 512, 512);
        k_wt<<<cdiv(512 * 512, 256), 256, 0, stream>>>(m.ch_w2, WT2, 512, 512);
        gemm_bf16<EPI_GELU_BF16><<<gCH, 256, 0, stream>>>(
            LNBF, WT1, m.ch_b1, nullptr, nullptr, H1BF, nullptr, nullptr, 34816, 512, 512);
        gemm_bf16<EPI_RES_F32><<<gCH, 256, 0, stream>>>(
            H1BF, WT2, m.ch_b2, XY, X, nullptr, nullptr, nullptr, 34816, 512, 512);
    }

    // --- 3. final encoder LN + joints->tokens + feat embed ---
    k_layernorm<<<(int)ROWS, 256, 0, stream>>>(X, enc_ln_g, enc_ln_b, LN1F, nullptr, 512);
    k_token_mix_km<17, 34><<<cdiv(2048 * 512, 256), 256, 0, stream>>>(
        LN1F, token_mlp_w, token_mlp_b, TOKBF, 2048, 512);
    k_wt<<<cdiv(512 * 512, 256), 256, 0, stream>>>(feat_embed_w, FEWT, 512, 512);
    gemm_bf16<EPI_BIAS_BF16><<<gFE, 256, 0, stream>>>(
        TOKBF, FEWT, feat_embed_b, nullptr, nullptr, FLATBF, nullptr, nullptr, 69632, 512, 512);

    // --- 4. VQ: fused distance GEMM + argmin ---
    k_cb_prep<<<2048, 256, 0, stream>>>(codebook, CBBF, CBN);
    k_init64<<<cdiv(69632, 256), 256, 0, stream>>>(ARGWS, 69632);
    gemm_bf16<EPI_VQ><<<gVQ, 256, 0, stream>>>(
        FLATBF, CBBF, nullptr, nullptr, nullptr, nullptr, CBN, ARGWS, 69632, 2048, 512);
    k_idx_out<<<cdiv(69632, 256), 256, 0, stream>>>(
        ARGWS, (int*)d_out + 69632, 69632);  // indices after 2048*17*2 floats

    // --- 5. decoder: gather + tokens->joints + dec_start GEMM ---
    k_dec_token_mix<<<cdiv(2048 * 512, 256), 256, 0, stream>>>(
        codebook, ARGWS, dec_tok_mlp_w, dec_tok_mlp_b, PJBF);
    k_wt<<<cdiv(512 * 32, 256), 256, 0, stream>>>(dec_start_w, DSWT, 512, 32);
    gemm_bf16<EPI_BIAS_F32><<<gDS, 256, 0, stream>>>(
        PJBF, DSWT, dec_start_b, nullptr, DEC, nullptr, nullptr, nullptr, 34816, 32, 512);

    // --- 6. decoder mixer block (H=32) ---
    k_layernorm<<<(int)ROWS, 256, 0, stream>>>(DEC, dec_mix.ln1_g, dec_mix.ln1_b, DECLN, nullptr, 32);
    k_token_mix<17, 64><<<cdiv(2048 * 32, 256), 256, 0, stream>>>(
        DECLN, DEC, dec_mix.tok_w1, dec_mix.tok_b1, dec_mix.tok_w2, dec_mix.tok_b2, DECXY, 2048, 32);
    k_layernorm<<<(int)ROWS, 256, 0, stream>>>(DECXY, dec_mix.ln2_g, dec_mix.ln2_b, DECLN, nullptr, 32);
    k_dec_channel<<<cdiv(34816, 256), 256, 0, stream>>>(
        DECLN, DECXY, dec_mix.ch_w1, dec_mix.ch_b1, dec_mix.ch_w2, dec_mix.ch_b2, DEC, 34816);

    // --- 7. final LN + recover -> d_out[0 : 2048*17*2] ---
    k_recover<<<cdiv(34816, 256), 256, 0, stream>>>(
        DEC, dec_ln_g, dec_ln_b, recover_w, recover_b, (float*)d_out, 34816);
}
